// MHA_34222299414725
// MI455X (gfx1250) — matrix operand order repaired and checked by reading, not compiled, not run on hardware
//
#include <hip/hip_runtime.h>

typedef unsigned short u16;
typedef __attribute__((ext_vector_type(16))) __bf16 bf16x16;
typedef __attribute__((ext_vector_type(8)))  float  f32x8;
typedef __attribute__((__vector_size__(4 * sizeof(int)))) int v4i_t;

__device__ __forceinline__ u16 f2bf(float f) {
    unsigned u = __builtin_bit_cast(unsigned, f);
    u += 0x7FFFu + ((u >> 16) & 1u);          // round-to-nearest-even
    return (u16)(u >> 16);
}
__device__ __forceinline__ float bf2f(u16 h) {
    unsigned u = ((unsigned)h) << 16;
    return __builtin_bit_cast(float, u);
}

union FragU { bf16x16 v; uint4 q[2]; };

// A-matrix (16x32 bf16, MxK): lane row = caller; chunks at K = (lane/16)*8 and +16
__device__ __forceinline__ bf16x16 load_frag_a(const u16* rowbase, int lh) {
    FragU u;
    const u16* p = rowbase + lh * 8;
    u.q[0] = *(const uint4*)(p);
    u.q[1] = *(const uint4*)(p + 16);
    return u.v;
}
// B-matrix (32x16 bf16, KxN): lane col = caller; chunks at K = (lane/16)*8 and +16
__device__ __forceinline__ bf16x16 load_frag_b(const u16* rowbase, int lh) {
    FragU u;
    const u16* p = rowbase + lh * 8;
    u.q[0] = *(const uint4*)(p);
    u.q[1] = *(const uint4*)(p + 16);
    return u.v;
}
__device__ __forceinline__ f32x8 wmma_bf16(bf16x16 a, bf16x16 b, f32x8 c) {
    return __builtin_amdgcn_wmma_f32_16x16x32_bf16(false, a, false, b, (short)0, c, false, false);
}

// ---------- CDNA5 async global->LDS copy (ASYNCcnt path), with sync fallback ----------
#if __has_builtin(__builtin_amdgcn_global_load_async_to_lds_b128)
__device__ __forceinline__ void async_cp16(u16* dst_lds, const u16* src_glb) {
    __builtin_amdgcn_global_load_async_to_lds_b128(
        (__attribute__((address_space(1))) v4i_t*)src_glb,
        (__attribute__((address_space(3))) v4i_t*)dst_lds, 0, 0);
}
__device__ __forceinline__ void async_wait0() {
#if __has_builtin(__builtin_amdgcn_s_wait_asynccnt)
    __builtin_amdgcn_s_wait_asynccnt(0);
#else
    asm volatile("s_wait_asynccnt 0x0" ::: "memory");
#endif
}
#else
__device__ __forceinline__ void async_cp16(u16* dst_lds, const u16* src_glb) {
    *(uint4*)dst_lds = *(const uint4*)src_glb;
}
__device__ __forceinline__ void async_wait0() {}
#endif

// ---------------- fp32 -> bf16 conversion ----------------
__global__ void cvt_f32_to_bf16(const float* __restrict__ src, u16* __restrict__ dst, long long n) {
    long long i = (long long)blockIdx.x * blockDim.x + threadIdx.x;
    long long stride = (long long)gridDim.x * blockDim.x;
    for (; i < n; i += stride) dst[i] = f2bf(src[i]);
}

// ---------------- bf16 GEMM: y[m,n] = sum_k A[m,k]*W[n,k] (+bias) ----------------
// 128x128 block tile, 8 waves of 32x64, K-step 32, double-buffered async LDS staging.
// mode 0: fp32 out + bias b0. mode 1: scatter bf16 to q/k/v [B,H,S,HD] with per-slab bias.
__global__ __launch_bounds__(256) void gemm_bf16(
    const u16* __restrict__ A, const u16* __restrict__ W,
    const float* __restrict__ b0, const float* __restrict__ b1, const float* __restrict__ b2,
    float* __restrict__ outf,
    u16* __restrict__ qo, u16* __restrict__ ko, u16* __restrict__ vo,
    int N, int mode)
{
    const int K = 1024;
    __shared__ u16 As[2][128 * 32];
    __shared__ u16 Bs[2][128 * 32];
    const int tid = threadIdx.x;
    const int m0 = blockIdx.y * 128;
    const int n0 = blockIdx.x * 128;
    const int lane = tid & 31, wv = tid >> 5;
    const int wm = wv & 3, wn = wv >> 2;          // 4 x 2 wave grid
    const int lr = lane & 15, lh = lane >> 4;

    // stage one K-step (128x32 of A and of W) into buffer `buf`
    auto stage = [&](int buf, int k0) {
        for (int i = 0; i < 2; ++i) {
            int idx = tid + 256 * i;              // 0..511 (512 16B chunks each)
            int row = idx >> 2, kc = (idx & 3) << 3;
            async_cp16(&As[buf][row * 32 + kc], &A[(size_t)(m0 + row) * K + k0 + kc]);
            async_cp16(&Bs[buf][row * 32 + kc], &W[(size_t)(n0 + row) * K + k0 + kc]);
        }
    };

    stage(0, 0);
    async_wait0();
    __syncthreads();

    f32x8 acc[2][4] = {};
    int buf = 0;
    for (int k0 = 0; k0 < K; k0 += 32) {
        if (k0 + 32 < K) stage(buf ^ 1, k0 + 32);   // prefetch next tile async
        bf16x16 af[2], bfr[4];
        for (int tm = 0; tm < 2; ++tm)
            af[tm] = load_frag_a(&As[buf][(32 * wm + 16 * tm + lr) * 32], lh);
        for (int tn = 0; tn < 4; ++tn)
            bfr[tn] = load_frag_b(&Bs[buf][(64 * wn + 16 * tn + lr) * 32], lh);
        for (int tm = 0; tm < 2; ++tm)
            for (int tn = 0; tn < 4; ++tn)
                acc[tm][tn] = wmma_bf16(af[tm], bfr[tn], acc[tm][tn]);
        async_wait0();
        __syncthreads();
        buf ^= 1;
    }

    if (mode == 0) {
        for (int tm = 0; tm < 2; ++tm)
            for (int tn = 0; tn < 4; ++tn) {
                int n = n0 + 64 * wn + 16 * tn + lr;
                float bv = b0[n];
                for (int r = 0; r < 8; ++r) {
                    int m = m0 + 32 * wm + 16 * tm + r + 8 * lh;
                    outf[(size_t)m * N + n] = acc[tm][tn][r] + bv;
                }
            }
    } else {
        const int which = n0 >> 10;               // uniform per block (128 | 1024)
        const float* bias = (which == 0) ? b0 : (which == 1) ? b1 : b2;
        u16* dst          = (which == 0) ? qo : (which == 1) ? ko : vo;
        const int nloc0 = n0 & 1023;
        for (int tm = 0; tm < 2; ++tm)
            for (int tn = 0; tn < 4; ++tn) {
                int nl = nloc0 + 64 * wn + 16 * tn + lr;
                float bv = bias[nl];
                int hh = nl >> 5, hd = nl & 31;
                for (int r = 0; r < 8; ++r) {
                    int m = m0 + 32 * wm + 16 * tm + r + 8 * lh;
                    int bI = m >> 6, s = m & 63;
                    size_t off = (((size_t)(bI * 32 + hh)) * 64 + s) * 32 + hd;
                    dst[off] = f2bf(acc[tm][tn][r] + bv);
                }
            }
    }
}

// ---------------- attention with RPE bias, one (b,h) per block ----------------
// q/k/v: [B*H][64][32] bf16. rpe_q/rpe_k: [1024][225] fp32. ctx: [B*64][1024] bf16.
__global__ __launch_bounds__(128) void attn_rpe(
    const u16* __restrict__ qb, const u16* __restrict__ kb, const u16* __restrict__ vb,
    const float* __restrict__ rpe_q, const float* __restrict__ rpe_k,
    u16* __restrict__ ctx)
{
    __shared__ u16 sm_qs[64 * 32];    // Q tile, row-major [s][d]
    __shared__ u16 sm_ks[64 * 32];    // K tile, row-major [s][d]
    __shared__ u16 sm_vt[32 * 64];    // V transposed [d][s]
    __shared__ u16 sm_rqT[240 * 32];  // RQ^T [r][d] (225 padded to 240)
    __shared__ u16 sm_rkT[240 * 32];  // RK^T [r][d]
    __shared__ u16 sm_tq[64 * 240];   // t_q[m][r] bf16
    __shared__ u16 sm_tk[64 * 240];   // t_k[n][r] bf16
    __shared__ u16 sm_ps[64 * 64];    // softmax probs bf16

    const int bh = blockIdx.x;        // b*32 + h
    const int h  = bh & 31;
    const int bI = bh >> 5;
    const int tid = threadIdx.x;
    const int lane = tid & 31, wv = tid >> 5;
    const int lr = lane & 15, lh = lane >> 4;

    const u16* qg = qb + (size_t)bh * 2048;
    const u16* kg = kb + (size_t)bh * 2048;
    const u16* vg = vb + (size_t)bh * 2048;

    // stage Q, K via async global->LDS (256 x 16B chunks each)
    for (int i = 0; i < 2; ++i) {
        int idx = tid + 128 * i;
        async_cp16(&sm_qs[idx * 8], &qg[idx * 8]);
        async_cp16(&sm_ks[idx * 8], &kg[idx * 8]);
    }
    // stage V transposed
    for (int i = 0; i < 16; ++i) {
        int e = tid + 128 * i;
        int key = e >> 5, d = e & 31;
        sm_vt[d * 64 + key] = vg[e];
    }
    // stage RPE tables transposed + bf16 (r along consecutive lanes -> coalesced)
    for (int i = 0; i < 60; ++i) {
        int e = tid + 128 * i;      // 0..7679
        int d = e / 240;
        int r = e - d * 240;
        u16 vq = 0, vk = 0;
        if (r < 225) {
            int gi = (d * 32 + h) * 225 + r;
            vq = f2bf(rpe_q[gi]);
            vk = f2bf(rpe_k[gi]);
        }
        sm_rqT[r * 32 + d] = vq;
        sm_rkT[r * 32 + d] = vk;
    }
    async_wait0();
    __syncthreads();

    // A fragments: wave wv owns rows 16*wv .. 16*wv+15
    bf16x16 aq = load_frag_a(&sm_qs[(16 * wv + lr) * 32], lh);
    bf16x16 ak = load_frag_a(&sm_ks[(16 * wv + lr) * 32], lh);

    // scores strip: S[16 x 64] = Q_strip @ K^T (K=32 in one WMMA per tile)
    f32x8 acc[4];
    for (int nt = 0; nt < 4; ++nt) {
        bf16x16 bk = load_frag_b(&sm_ks[(16 * nt + lr) * 32], lh);
        f32x8 z = {};
        acc[nt] = wmma_bf16(aq, bk, z);
    }
    // t_q strip = Q_strip @ RQ (32x240), t_k strip = K_strip @ RK
    for (int rc = 0; rc < 15; ++rc) {
        bf16x16 brq = load_frag_b(&sm_rqT[(16 * rc + lr) * 32], lh);
        bf16x16 brk = load_frag_b(&sm_rkT[(16 * rc + lr) * 32], lh);
        f32x8 z0 = {}, z1 = {};
        f32x8 tq = wmma_bf16(aq, brq, z0);
        f32x8 tk = wmma_bf16(ak, brk, z1);
        for (int r = 0; r < 8; ++r) {
            int row = 16 * wv + r + 8 * lh;
            int col = 16 * rc + lr;
            sm_tq[row * 240 + col] = f2bf(tq[r]);
            sm_tk[row * 240 + col] = f2bf(tk[r]);
        }
    }
    __syncthreads();

    // bias gather + scale
    const float sc = 0.17677669529663687f;  // 1/sqrt(32)
    for (int nt = 0; nt < 4; ++nt)
        for (int r = 0; r < 8; ++r) {
            int m = 16 * wv + r + 8 * lh;
            int n = 16 * nt + lr;
            int rel = 15 * ((m >> 3) - (n >> 3) + 7) + ((m & 7) - (n & 7) + 7);
            float vq = bf2f(sm_tq[m * 240 + rel]);
            float vk = bf2f(sm_tk[n * 240 + rel]);
            acc[nt][r] = (acc[nt][r] + vq + vk) * sc;
        }

    // softmax: each row m lives in one 16-lane half across 4 accum tiles
    for (int r = 0; r < 8; ++r) {
        float mx = fmaxf(fmaxf(acc[0][r], acc[1][r]), fmaxf(acc[2][r], acc[3][r]));
        for (int d = 1; d < 16; d <<= 1) mx = fmaxf(mx, __shfl_xor(mx, d));
        float s = 0.f;
        for (int nt = 0; nt < 4; ++nt) {
            float e = __expf(acc[nt][r] - mx);
            acc[nt][r] = e;
            s += e;
        }
        for (int d = 1; d < 16; d <<= 1) s += __shfl_xor(s, d);
        float inv = 1.0f / s;
        int m = 16 * wv + r + 8 * lh;
        for (int nt = 0; nt < 4; ++nt)
            sm_ps[m * 64 + 16 * nt + lr] = f2bf(acc[nt][r] * inv);
    }
    __syncthreads();

    // O_strip[16 x 32] = P_strip[16 x 64] @ V[64 x 32]  (two K=32 WMMAs per N-tile)
    bf16x16 a0 = load_frag_a(&sm_ps[(16 * wv + lr) * 64], lh);
    bf16x16 a1 = load_frag_a(&sm_ps[(16 * wv + lr) * 64 + 32], lh);
    for (int nt = 0; nt < 2; ++nt) {
        bf16x16 b0 = load_frag_b(&sm_vt[(16 * nt + lr) * 64], lh);
        bf16x16 b1 = load_frag_b(&sm_vt[(16 * nt + lr) * 64 + 32], lh);
        f32x8 z = {};
        f32x8 o = wmma_bf16(a0, b0, z);
        o = wmma_bf16(a1, b1, o);
        for (int r = 0; r < 8; ++r) {
            int m = 16 * wv + r + 8 * lh;
            int n = 16 * nt + lr;
            size_t row = (size_t)bI * 64 + m;
            ctx[row * 1024 + h * 32 + n] = f2bf(o[r]);
        }
    }
}

extern "C" void kernel_launch(void* const* d_in, const int* in_sizes, int n_in,
                              void* d_out, int out_size, void* d_ws, size_t ws_size,
                              hipStream_t stream) {
    (void)in_sizes; (void)n_in; (void)out_size; (void)ws_size;
    const float* x     = (const float*)d_in[0];
    const float* q_w   = (const float*)d_in[1];
    const float* q_b   = (const float*)d_in[2];
    const float* k_w   = (const float*)d_in[3];
    const float* k_b   = (const float*)d_in[4];
    const float* v_w   = (const float*)d_in[5];
    const float* v_b   = (const float*)d_in[6];
    const float* out_w = (const float*)d_in[7];
    const float* out_b = (const float*)d_in[8];
    const float* rpe_q = (const float*)d_in[9];
    const float* rpe_k = (const float*)d_in[10];

    char* ws = (char*)d_ws;
    u16* xb   = (u16*)(ws);                         // 65536*1024 bf16 = 128 MB
    u16* qbuf = (u16*)(ws + 134217728ull);          // 128 MB
    u16* kbuf = (u16*)(ws + 268435456ull);          // 128 MB
    u16* vbuf = (u16*)(ws + 402653184ull);          // 128 MB
    u16* ctx  = (u16*)(ws + 536870912ull);          // 128 MB
    u16* wqkv = (u16*)(ws + 671088640ull);          // 3072*1024 bf16 = 6 MB
    u16* wout = (u16*)(ws + 677380096ull);          // 1024*1024 bf16 = 2 MB

    cvt_f32_to_bf16<<<8192, 256, 0, stream>>>(x, xb, 67108864LL);
    cvt_f32_to_bf16<<<1024, 256, 0, stream>>>(q_w, wqkv, 1048576LL);
    cvt_f32_to_bf16<<<1024, 256, 0, stream>>>(k_w, wqkv + 1048576, 1048576LL);
    cvt_f32_to_bf16<<<1024, 256, 0, stream>>>(v_w, wqkv + 2097152, 1048576LL);
    cvt_f32_to_bf16<<<1024, 256, 0, stream>>>(out_w, wout, 1048576LL);

    gemm_bf16<<<dim3(24, 512), 256, 0, stream>>>(xb, wqkv, q_b, k_b, v_b,
                                                 nullptr, qbuf, kbuf, vbuf, 3072, 1);
    attn_rpe<<<32768, 128, 0, stream>>>(qbuf, kbuf, vbuf, rpe_q, rpe_k, ctx);
    gemm_bf16<<<dim3(8, 512), 256, 0, stream>>>(ctx, wout, out_b, out_b, out_b,
                                                (float*)d_out, nullptr, nullptr, nullptr, 1024, 0);
}
